// Head_31593779429737
// MI455X (gfx1250) — matrix operand order repaired and checked by reading, not compiled, not run on hardware
//
#include <hip/hip_runtime.h>

#define B_ 8
#define S_ 2048
#define E_ 1024
#define H_ 128

typedef __attribute__((ext_vector_type(16))) __bf16 v16bf;
typedef __attribute__((ext_vector_type(8)))  float  v8f;
typedef int v4i __attribute__((__vector_size__(16)));

// One gfx1250 async DMA: 16 bytes (8 bf16) global -> LDS, tracked by ASYNCcnt.
__device__ __forceinline__ void async_b128(const __bf16* g, __bf16* l) {
    __builtin_amdgcn_global_load_async_to_lds_b128(
        (__attribute__((address_space(1))) v4i*)(g),
        (__attribute__((address_space(3))) v4i*)(l),
        0, 0);
}
__device__ __forceinline__ void wait_async0() {
    asm volatile("s_wait_asynccnt 0" ::: "memory");
}

// A-matrix 16-bit fragment (16xK=32): per-lane K pattern {0-7,16-23} / {8-15,24-31}.
// p must point at (row base + khalf) where khalf = (lane>>4)*8 elements.
__device__ __forceinline__ v16bf load_fragA(const __bf16* p) {
    union { v16bf v; uint4 u[2]; } t;
    t.u[0] = *(const uint4*)(p);
    t.u[1] = *(const uint4*)(p + 16);
    return t.v;
}
// B-matrix 16-bit fragment (K=32 x 16): per-lane K pattern {0-7,16-23} / {8-15,24-31}.
// p must point at (column base + (lane>>4)*8 elements).
__device__ __forceinline__ v16bf load_fragB(const __bf16* p) {
    union { v16bf v; uint4 u[2]; } t;
    t.u[0] = *(const uint4*)(p);
    t.u[1] = *(const uint4*)(p + 16);
    return t.v;
}

// ---------------------------------------------------------------------------
// Kernel 1: convert Wq/Wk/Wv fp32 [E][H] row-major -> bf16 column-major [H][E]
// ---------------------------------------------------------------------------
__global__ __launch_bounds__(256) void cvt_w_kernel(const float* __restrict__ Wq,
                                                    const float* __restrict__ Wk,
                                                    const float* __restrict__ Wv,
                                                    __bf16* __restrict__ Wb) {
    int idx = blockIdx.x * blockDim.x + threadIdx.x;      // 0 .. 3*E*H-1
    int w   = idx / (E_ * H_);
    int rem = idx % (E_ * H_);
    int e = rem / H_;
    int h = rem % H_;
    const float* W = (w == 0) ? Wq : (w == 1) ? Wk : Wv;
    Wb[(size_t)w * E_ * H_ + (size_t)h * E_ + e] = (__bf16)W[(size_t)e * H_ + h];
}

// ---------------------------------------------------------------------------
// Kernel 2: projection GEMM  dst[b,s,:] = X[b,s,:] @ W + bias   (bf16 out)
// grid.x = (B*S)/64 row tiles, grid.y = 3 (q,k,v); 128 threads = 4 waves.
// W K-slices (128 cols x 32 K = 8KB) staged in LDS via async DMA, double
// buffered so the next slice's DMA overlaps the current 8 WMMAs.  B-fragment
// loads are software-pipelined (depth 1) so ds_load latency hides under WMMA.
// ---------------------------------------------------------------------------
__global__ __launch_bounds__(128) void proj_kernel(const float* __restrict__ Q,
                                                   const float* __restrict__ K,
                                                   const float* __restrict__ V,
                                                   const float* __restrict__ bq,
                                                   const float* __restrict__ bk,
                                                   const float* __restrict__ bv,
                                                   const __bf16* __restrict__ Wb_all,
                                                   __bf16* __restrict__ qb,
                                                   __bf16* __restrict__ kb,
                                                   __bf16* __restrict__ vb) {
    __shared__ __bf16 lds_w[2][128 * 32];   // [buf][col][k]  2 x 8KB

    const int which = blockIdx.y;
    const float*  X    = (which == 0) ? Q  : (which == 1) ? K  : V;
    const float*  bias = (which == 0) ? bq : (which == 1) ? bk : bv;
    const __bf16* Wb   = Wb_all + (size_t)which * E_ * H_;
    __bf16*       dst  = (which == 0) ? qb : (which == 1) ? kb : vb;

    const int tid    = threadIdx.x;
    const int wave   = tid >> 5;
    const int lane   = tid & 31;
    const int lane16 = lane & 15;
    const int half   = lane >> 4;
    const int khalf  = half * 8;

    const int row0 = blockIdx.x * 64 + wave * 16;   // first row of this wave
    const int r    = row0 + lane16;                 // this lane's A row

    v8f acc[8];
#pragma unroll
    for (int i = 0; i < 8; ++i) acc[i] = (v8f){};

    // Prologue: async-stage W K-slice 0 (thread t copies column t, 64B).
    {
        const __bf16* g = Wb + (size_t)tid * E_;
        __bf16* l = lds_w[0] + tid * 32;
#pragma unroll
        for (int i = 0; i < 4; ++i) async_b128(g + i * 8, l + i * 8);
    }
    wait_async0();
    __syncthreads();

    const int NK = E_ / 32;
    for (int ks = 0; ks < NK; ++ks) {
        const int cur = ks & 1;
        const int k0  = ks * 32;

        // Kick off DMA of the next W K-slice into the other buffer.
        if (ks + 1 < NK) {
            const __bf16* g = Wb + (size_t)tid * E_ + (k0 + 32);
            __bf16* l = lds_w[cur ^ 1] + tid * 32;
#pragma unroll
            for (int i = 0; i < 4; ++i) async_b128(g + i * 8, l + i * 8);
        }

        // A fragment: X row (fp32) -> bf16, pattern {khalf..+7, khalf+16..+23}
        const float* xp = X + (size_t)r * E_ + k0 + khalf;
        float4 x0 = *(const float4*)(xp);
        float4 x1 = *(const float4*)(xp + 4);
        float4 x2 = *(const float4*)(xp + 16);
        float4 x3 = *(const float4*)(xp + 20);
        v16bf a;
        a[0]  = (__bf16)x0.x; a[1]  = (__bf16)x0.y; a[2]  = (__bf16)x0.z; a[3]  = (__bf16)x0.w;
        a[4]  = (__bf16)x1.x; a[5]  = (__bf16)x1.y; a[6]  = (__bf16)x1.z; a[7]  = (__bf16)x1.w;
        a[8]  = (__bf16)x2.x; a[9]  = (__bf16)x2.y; a[10] = (__bf16)x2.z; a[11] = (__bf16)x2.w;
        a[12] = (__bf16)x3.x; a[13] = (__bf16)x3.y; a[14] = (__bf16)x3.z; a[15] = (__bf16)x3.w;

        // 8 WMMAs with depth-1 pipelined B-fragment loads.
        const __bf16* wbase = lds_w[cur] + half * 8;
        v16bf bcur = load_fragB(wbase + (lane16) * 32);
#pragma unroll
        for (int nt = 0; nt < 8; ++nt) {
            v16bf bnext = bcur;
            if (nt + 1 < 8)
                bnext = load_fragB(wbase + ((nt + 1) * 16 + lane16) * 32);
            acc[nt] = __builtin_amdgcn_wmma_f32_16x16x32_bf16(
                false, a, false, bcur, (short)0, acc[nt], false, false);
            bcur = bnext;
        }

        if (ks + 1 < NK) wait_async0();
        __syncthreads();
    }

    // Epilogue: + bias, cast to bf16, store.  C layout: M = j + 8*half, N = lane16.
#pragma unroll
    for (int nt = 0; nt < 8; ++nt) {
        int h = nt * 16 + lane16;
        float bb = bias[h];
#pragma unroll
        for (int j = 0; j < 8; ++j) {
            int rr = row0 + j + half * 8;
            dst[(size_t)rr * H_ + h] = (__bf16)(acc[nt][j] + bb);
        }
    }
}

// ---------------------------------------------------------------------------
// Kernel 3: causal flash attention.
// grid = B * (S/32); 64 threads = 2 waves; wave w owns rows qbase+16w..+15.
// Key blocks of 32: K staged row-major via async DMA (double buffered),
// V staged transposed via a software-pipelined register round-trip.
// ---------------------------------------------------------------------------
__global__ __launch_bounds__(64) void attn_kernel(const __bf16* __restrict__ qb,
                                                  const __bf16* __restrict__ kb,
                                                  const __bf16* __restrict__ vb,
                                                  float* __restrict__ out) {
    __shared__ __bf16 lds_k[2][32 * H_];    // [buf][key][h]   2 x 8 KB
    __shared__ __bf16 lds_vt[2][H_ * 32];   // [buf][h][key]   2 x 8 KB
    __shared__ __bf16 lds_p[2][16 * 32];    // per-wave P      2 KB

    const int nqt   = S_ / 32;
    const int b     = blockIdx.x / nqt;
    const int qtile = blockIdx.x % nqt;
    const int qbase = qtile * 32;

    const int tid    = threadIdx.x;
    const int wave   = tid >> 5;
    const int lane   = tid & 31;
    const int lane16 = lane & 15;
    const int half   = lane >> 4;
    const int khalf  = half * 8;

    const float scale = 0.08838834764831845f;   // 1/sqrt(128)

    // V staging assignment: thread t handles key=t>>1, h segment of 64.
    const int vkey = tid >> 1;
    const int vseg = tid & 1;

    // Load this wave's Q fragments (4 chunks of K=32 over H=128), already bf16.
    const int qrow = qbase + wave * 16 + lane16;
    v16bf qf[4];
#pragma unroll
    for (int c = 0; c < 4; ++c)
        qf[c] = load_fragA(qb + ((size_t)(b * S_ + qrow)) * H_ + c * 32 + khalf);

    v8f O[8];
#pragma unroll
    for (int i = 0; i < 8; ++i) O[i] = (v8f){};
    float m[8], l[8];
#pragma unroll
    for (int j = 0; j < 8; ++j) { m[j] = -1e30f; l[j] = 0.0f; }

    union VStage { uint4 u[8]; __bf16 e[64]; };

    // ---- prologue: stage key block 0 into buffer 0 ----
    {
        const __bf16* kg = kb + ((size_t)(b * S_)) * H_ + tid * 64;
        __bf16* kl = lds_k[0] + tid * 64;
#pragma unroll
        for (int i = 0; i < 8; ++i) async_b128(kg + i * 8, kl + i * 8);

        VStage vs;
        const uint4* vg = (const uint4*)(vb + ((size_t)(b * S_ + vkey)) * H_ + vseg * 64);
#pragma unroll
        for (int i = 0; i < 8; ++i) vs.u[i] = vg[i];
#pragma unroll
        for (int i = 0; i < 64; ++i)
            lds_vt[0][(vseg * 64 + i) * 32 + vkey] = vs.e[i];
    }
    wait_async0();
    __syncthreads();

    const int nkb = qtile + 1;                  // causal upper bound
    for (int kbi = 0; kbi < nkb; ++kbi) {
        const int kbase = kbi * 32;
        const int cur   = kbi & 1;

        // ---- kick off staging of the NEXT key block (DMA + V prefetch) ----
        VStage vs;
        const bool more = (kbi + 1) < nkb;
        if (more) {
            const int nb = kbase + 32;
            const __bf16* kg = kb + ((size_t)(b * S_ + nb)) * H_ + tid * 64;
            __bf16* kl = lds_k[cur ^ 1] + tid * 64;
#pragma unroll
            for (int i = 0; i < 8; ++i) async_b128(kg + i * 8, kl + i * 8);
            const uint4* vg = (const uint4*)(vb + ((size_t)(b * S_ + nb + vkey)) * H_ + vseg * 64);
#pragma unroll
            for (int i = 0; i < 8; ++i) vs.u[i] = vg[i];
        }

        // ---- scores: two 16x16 tiles, 8 WMMAs with pipelined K fragments ----
        // tile i: c = i>>1 (H chunk), odd i -> key half 1.
        const __bf16* kbp = lds_k[cur] + half * 8;
        v8f s0 = (v8f){}, s1 = (v8f){};
        v16bf kcur = load_fragB(kbp + (lane16) * H_);      // i = 0
#pragma unroll
        for (int i = 0; i < 8; ++i) {
            v16bf knext = kcur;
            if (i + 1 < 8) {
                int c2 = (i + 1) >> 1, h2 = (i + 1) & 1;
                knext = load_fragB(kbp + (h2 * 16 + lane16) * H_ + c2 * 32);
            }
            int c = i >> 1;
            if (i & 1)
                s1 = __builtin_amdgcn_wmma_f32_16x16x32_bf16(false, qf[c], false, kcur,
                                                             (short)0, s1, false, false);
            else
                s0 = __builtin_amdgcn_wmma_f32_16x16x32_bf16(false, qf[c], false, kcur,
                                                             (short)0, s0, false, false);
            kcur = knext;
        }

        // ---- online softmax (rows live across the 16 lanes of a half-wave) ----
        const int col0 = kbase + lane16;
#pragma unroll
        for (int j = 0; j < 8; ++j) {
            int rg = qbase + wave * 16 + j + khalf;     // global q row of element j
            float v0 = (col0      <= rg) ? s0[j] * scale : -1e30f;
            float v1 = (col0 + 16 <= rg) ? s1[j] * scale : -1e30f;
            float tm = fmaxf(v0, v1);
            tm = fmaxf(tm, __shfl_xor(tm, 1, 32));
            tm = fmaxf(tm, __shfl_xor(tm, 2, 32));
            tm = fmaxf(tm, __shfl_xor(tm, 4, 32));
            tm = fmaxf(tm, __shfl_xor(tm, 8, 32));
            float mn    = fmaxf(m[j], tm);
            float alpha = __expf(m[j] - mn);
            float p0 = __expf(v0 - mn);
            float p1 = __expf(v1 - mn);
            float rs = p0 + p1;
            rs += __shfl_xor(rs, 1, 32);
            rs += __shfl_xor(rs, 2, 32);
            rs += __shfl_xor(rs, 4, 32);
            rs += __shfl_xor(rs, 8, 32);
            l[j] = l[j] * alpha + rs;
            m[j] = mn;
#pragma unroll
            for (int t8 = 0; t8 < 8; ++t8) O[t8][j] *= alpha;
            // C layout -> LDS row-major P tile [row][key]
            lds_p[wave][(j + khalf) * 32 + lane16]      = (__bf16)p0;
            lds_p[wave][(j + khalf) * 32 + 16 + lane16] = (__bf16)p1;
        }

        // ---- P @ V : P as A fragment, V^T columns as pipelined B fragments ----
        v16bf pf = load_fragA(lds_p[wave] + lane16 * 32 + khalf);
        const __bf16* vbp = lds_vt[cur] + half * 8;
        v16bf vcur = load_fragB(vbp + (lane16) * 32);      // t8 = 0
#pragma unroll
        for (int t8 = 0; t8 < 8; ++t8) {
            v16bf vnext = vcur;
            if (t8 + 1 < 8)
                vnext = load_fragB(vbp + ((t8 + 1) * 16 + lane16) * 32);
            O[t8] = __builtin_amdgcn_wmma_f32_16x16x32_bf16(false, pf, false, vcur,
                                                            (short)0, O[t8], false, false);
            vcur = vnext;
        }

        // ---- retire next-block staging: V transpose scatter + async wait ----
        if (more) {
#pragma unroll
            for (int i = 0; i < 64; ++i)
                lds_vt[cur ^ 1][(vseg * 64 + i) * 32 + vkey] = vs.e[i];
            wait_async0();
        }
        __syncthreads();
    }

    // ---- normalize and store fp32 output ----
#pragma unroll
    for (int t8 = 0; t8 < 8; ++t8) {
        int h = t8 * 16 + lane16;
#pragma unroll
        for (int j = 0; j < 8; ++j) {
            int rg = qbase + wave * 16 + j + khalf;
            out[((size_t)(b * S_ + rg)) * H_ + h] = O[t8][j] / l[j];
        }
    }
}

// ---------------------------------------------------------------------------
extern "C" void kernel_launch(void* const* d_in, const int* in_sizes, int n_in,
                              void* d_out, int out_size, void* d_ws, size_t ws_size,
                              hipStream_t stream) {
    const float* Q  = (const float*)d_in[0];
    const float* K  = (const float*)d_in[1];
    const float* V  = (const float*)d_in[2];
    // d_in[3] = causal mask (implemented analytically)
    const float* Wq = (const float*)d_in[4];
    const float* bq = (const float*)d_in[5];
    const float* Wk = (const float*)d_in[6];
    const float* bk = (const float*)d_in[7];
    const float* Wv = (const float*)d_in[8];
    const float* bv = (const float*)d_in[9];

    const size_t BSH = (size_t)B_ * S_ * H_;
    __bf16* qb = (__bf16*)d_ws;
    __bf16* kb = qb + BSH;
    __bf16* vb = kb + BSH;
    __bf16* Wb = vb + BSH;                     // 3 * E*H bf16, column-major

    // 1) weights fp32 -> bf16 (transposed to column-major)
    cvt_w_kernel<<<(3 * E_ * H_) / 256, 256, 0, stream>>>(Wq, Wk, Wv, Wb);

    // 2) q/k/v projections (WMMA bf16, async-DMA W staging)
    proj_kernel<<<dim3((B_ * S_) / 64, 3), 128, 0, stream>>>(
        Q, K, V, bq, bk, bv, Wb, qb, kb, vb);

    // 3) causal flash attention (WMMA bf16, async-DMA K staging, online softmax)
    attn_kernel<<<B_ * (S_ / 32), 64, 0, stream>>>(qb, kb, vb, (float*)d_out);
}
